// Epipolar_Attention_11123965296974
// MI455X (gfx1250) — matrix operand order repaired and checked by reading, not compiled, not run on hardware
//
#include <hip/hip_runtime.h>
#include <math.h>

typedef _Float16 h16;
typedef __attribute__((ext_vector_type(16))) _Float16 v16h;
typedef __attribute__((ext_vector_type(8)))  float    v8f;
typedef __attribute__((ext_vector_type(4)))  unsigned u32x4;
typedef __attribute__((ext_vector_type(4)))  float    f32x4;

#define N_HEADS 12
#define DHEAD   64
#define EDIM    768
#define NTOK    1024
#define BATCH   4
#define MROWS   (BATCH*NTOK)   /* 4096 */
#define FFDIM   1536

// ---------------- WMMA fragment loaders (wave32, contiguous k order used by this code for A and B) ----------------
// A-frag (16x32 f16): lane L -> row = L%16; VGPR r holds K{kb+2r,kb+2r+1}
// with kb = (L/16)*16 within the 32-wide window.
__device__ __forceinline__ v16h lds_a_frag(const h16* base, int row, int stride, int kwin, int hh) {
    union { v16h v; unsigned u[8]; } r;
    const unsigned* p = (const unsigned*)(base + row * stride);
    int d0 = (kwin + hh * 16) >> 1;
    r.u[0] = p[d0 + 0];  r.u[1] = p[d0 + 1];  r.u[2] = p[d0 + 2];  r.u[3] = p[d0 + 3];
    r.u[4] = p[d0 + 4];  r.u[5] = p[d0 + 5];  r.u[6] = p[d0 + 6];  r.u[7] = p[d0 + 7];
    return r.v;
}
// B-frag (32x16 f16): lane L -> col = L%16; holds 16 consecutive K starting at
// kwin + (L/16)*16, read from a row of the transposed operand.
__device__ __forceinline__ v16h lds_b_frag(const h16* base, int row, int stride, int kwin, int hh) {
    union { v16h v; u32x4 q[2]; } r;
    const u32x4* p = (const u32x4*)(base + row * stride + kwin + hh * 16);
    r.q[0] = p[0];
    r.q[1] = p[1];
    return r.v;
}

// ---------------- generic tiled GEMM: C(MxN) = A(MxK,f16) * Bt(NxK,f16)^T + bias --------
#define EPI_SPLIT_QK 0   // f16 out at (B,H,N,64)
#define EPI_SPLIT_VT 1   // f16 out at (B,H,64,N)
#define EPI_F32      2   // f32 out row-major
#define EPI_GELU16   3   // gelu, f16 out row-major
#define EPI_RES32    4   // + resid, f32 out row-major

__global__ __launch_bounds__(256) void gemm_ws(
    const h16* __restrict__ A, const h16* __restrict__ Bt,
    const float* __restrict__ bias, const float* __restrict__ resid,
    void* __restrict__ out, int K, int Ncols, int mode)
{
    __shared__ h16 As[64 * 40];
    __shared__ h16 Bs[128 * 40];
    const int tid  = threadIdx.x;
    const int wave = tid >> 5, lane = tid & 31, hh = lane >> 4, l16 = lane & 15;
    const int mi = wave & 3, ni = wave >> 2;
    const int m0 = blockIdx.x * 64;
    const int n0 = blockIdx.y * 128;

    v8f acc[4];
#pragma unroll
    for (int t = 0; t < 4; ++t)
#pragma unroll
        for (int r = 0; r < 8; ++r) acc[t][r] = 0.0f;

    {
        int rowA = tid >> 2, cA = tid & 3;
        int rowB0 = (tid * 2) >> 2;  (void)rowB0;
        for (int k0 = 0; k0 < K; k0 += 32) {
            // A tile: 64 rows x 32 halves
            const h16* srcA = A + (size_t)(m0 + rowA) * K + k0 + cA * 8;
            *(u32x4*)&As[rowA * 40 + cA * 8] = *(const u32x4*)srcA;
            if (k0 + 32 < K) __builtin_prefetch(srcA + 32, 0, 1);
            // B tile: 128 rows x 32 halves (rows of Wt / K-major operand)
#pragma unroll
            for (int j = 0; j < 2; ++j) {
                int idx = tid + j * 256;
                int row = idx >> 2, c = idx & 3;
                const h16* srcB = Bt + (size_t)(n0 + row) * K + k0 + c * 8;
                *(u32x4*)&Bs[row * 40 + c * 8] = *(const u32x4*)srcB;
                if (k0 + 32 < K) __builtin_prefetch(srcB + 32, 0, 1);
            }
            __syncthreads();
            v16h af = lds_a_frag(As, mi * 16 + l16, 40, 0, hh);
#pragma unroll
            for (int nt = 0; nt < 4; ++nt) {
                v16h bf = lds_b_frag(Bs, ni * 64 + nt * 16 + l16, 40, 0, hh);
                acc[nt] = __builtin_amdgcn_wmma_f32_16x16x32_f16(
                    false, af, false, bf, (short)0, acc[nt], false, false);
            }
            __syncthreads();
        }
    }

#pragma unroll
    for (int nt = 0; nt < 4; ++nt) {
        int col = n0 + ni * 64 + nt * 16 + l16;
        float bv = bias[col];
#pragma unroll
        for (int r = 0; r < 8; ++r) {
            int row = m0 + mi * 16 + hh * 8 + r;
            float v = acc[nt][r] + bv;
            if (mode == EPI_SPLIT_QK) {
                int b = row >> 10, n = row & 1023, h = col >> 6, d = col & 63;
                ((h16*)out)[((size_t)(b * N_HEADS + h) * NTOK + n) * DHEAD + d] = (h16)v;
            } else if (mode == EPI_SPLIT_VT) {
                int b = row >> 10, n = row & 1023, h = col >> 6, d = col & 63;
                ((h16*)out)[((size_t)(b * N_HEADS + h) * DHEAD + d) * NTOK + n] = (h16)v;
            } else if (mode == EPI_F32) {
                ((float*)out)[(size_t)row * Ncols + col] = v;
            } else if (mode == EPI_GELU16) {
                float g = 0.5f * v * (1.0f + erff(v * 0.70710678118f));
                ((h16*)out)[(size_t)row * Ncols + col] = (h16)g;
            } else { // EPI_RES32
                ((float*)out)[(size_t)row * Ncols + col] =
                    v + resid[(size_t)row * Ncols + col];
            }
        }
    }
}

// ---------------- LayerNorm over last dim (C) --------------------------------
__global__ __launch_bounds__(256) void ln_rows(
    const float* __restrict__ in, const float* __restrict__ g,
    const float* __restrict__ b, float* __restrict__ of32,
    h16* __restrict__ of16, int C)
{
    __shared__ float red[256];
    int row = blockIdx.x;
    const float* x = in + (size_t)row * C;
    float s = 0.0f;
    for (int c = threadIdx.x; c < C; c += 256) s += x[c];
    red[threadIdx.x] = s; __syncthreads();
    for (int o = 128; o > 0; o >>= 1) {
        if (threadIdx.x < o) red[threadIdx.x] += red[threadIdx.x + o];
        __syncthreads();
    }
    float mu = red[0] / (float)C;
    __syncthreads();
    float s2 = 0.0f;
    for (int c = threadIdx.x; c < C; c += 256) { float d = x[c] - mu; s2 += d * d; }
    red[threadIdx.x] = s2; __syncthreads();
    for (int o = 128; o > 0; o >>= 1) {
        if (threadIdx.x < o) red[threadIdx.x] += red[threadIdx.x + o];
        __syncthreads();
    }
    float rstd = rsqrtf(red[0] / (float)C + 1e-5f);
    for (int c = threadIdx.x; c < C; c += 256) {
        float y = (x[c] - mu) * rstd * g[c] + b[c];
        if (of32) of32[(size_t)row * C + c] = y;
        if (of16) of16[(size_t)row * C + c] = (h16)y;
    }
}

// ---------------- weight transpose + f32->f16 convert ------------------------
__global__ __launch_bounds__(256) void transpose_w(
    const float* __restrict__ in, h16* __restrict__ out, int K, int Ncols)
{
    __shared__ float t[32][33];
    int k0 = blockIdx.x * 32, n0 = blockIdx.y * 32;
    int tx = threadIdx.x & 31, ty = threadIdx.x >> 5;
#pragma unroll
    for (int j = 0; j < 4; ++j) {
        int r = ty + j * 8;
        t[r][tx] = in[(size_t)(k0 + r) * Ncols + n0 + tx];
    }
    __syncthreads();
#pragma unroll
    for (int j = 0; j < 4; ++j) {
        int r = ty + j * 8;
        out[(size_t)(n0 + r) * K + k0 + tx] = (h16)t[tx][r];
    }
}

// ---------------- epipolar geometry ------------------------------------------
struct Op4 { float x, y, z, w; };

__device__ __forceinline__ void mat3_inv(const float m[9], float inv[9]) {
    float a=m[0],b=m[1],c=m[2],d=m[3],e=m[4],f=m[5],g=m[6],h=m[7],i=m[8];
    float A = e*i - f*h, B = -(d*i - f*g), C = d*h - e*g;
    float id = 1.0f / (a*A + b*B + c*C);
    inv[0]=A*id;             inv[1]=-(b*i-c*h)*id;  inv[2]=(b*f-c*e)*id;
    inv[3]=B*id;             inv[4]=(a*i-c*g)*id;   inv[5]=-(a*f-c*d)*id;
    inv[6]=C*id;             inv[7]=-(a*h-b*g)*id;  inv[8]=(a*e-b*d)*id;
}
__device__ __forceinline__ void mv3(const float m[9], const float v[3], float o[3]) {
    o[0]=m[0]*v[0]+m[1]*v[1]+m[2]*v[2];
    o[1]=m[3]*v[0]+m[4]*v[1]+m[5]*v[2];
    o[2]=m[6]*v[0]+m[7]*v[1]+m[8]*v[2];
}
__device__ __forceinline__ float epi_dw(const Op4 op, float oij0, float oij1,
                                        float mx, float my) {
    float ocx = mx - oij0, ocy = my - oij1;             // oc.z == 0
    float crx = -op.z * ocy, cry = op.z * ocx, crz = op.x * ocy - op.y * ocx;
    float dist = sqrtf(crx*crx + cry*cry + crz*crz) * op.w;
    float sg = 1.0f / (1.0f + __expf(-50.0f * (dist - 0.5f)));
    return 1.0f - sg;
}

// grid = 2*B blocks; dir0: src=c2w[1],tgt=c2w[0]; dir1: src=c2w[0],tgt=c2w[1]
__global__ __launch_bounds__(128) void geom_prep(
    const float* __restrict__ intr, const float* __restrict__ c2w,
    float* __restrict__ opi, float* __restrict__ oijb)
{
    int dir = blockIdx.x >> 2, b = blockIdx.x & 3;
    int si = (dir == 0) ? 1 : 0, ti = 1 - si;
    const float* Kp = intr + b * 16;
    float k3[9] = { Kp[0],Kp[1],Kp[2], Kp[4],Kp[5],Kp[6], Kp[8],Kp[9],Kp[10] };
    const float Wf = 32.0f * 16.0f / 9.0f;
    k3[0]*=Wf; k3[1]*=Wf; k3[2]*=Wf;
    k3[3]*=32.f; k3[4]*=32.f; k3[5]*=32.f;
    k3[2]=16.f; k3[5]=16.f;
    const float* S = c2w + ((size_t)si * BATCH + b) * 16;
    const float* T = c2w + ((size_t)ti * BATCH + b) * 16;
    float sr[9] = { S[0],S[1],S[2], S[4],S[5],S[6], S[8],S[9],S[10] };
    float st[3] = { S[3], S[7], S[11] };
    float tr[9] = { T[0],T[1],T[2], T[4],T[5],T[6], T[8],T[9],T[10] };
    float tt[3] = { T[3], T[7], T[11] };
    float ri[9]; mat3_inv(tr, ri);
    float o2[3]; mv3(ri, st, o2);
    o2[0]-=tt[0]; o2[1]-=tt[1]; o2[2]-=tt[2];
    float oj[3]; mv3(k3, o2, oj);
    float oij0 = oj[0] / oj[2], oij1 = oj[1] / oj[2];
    if (threadIdx.x == 0) {
        float* o = oijb + (dir * BATCH + b) * 4;
        o[0]=oij0; o[1]=oij1; o[2]=1.0f; o[3]=0.0f;
    }
    float fx=k3[0], fy=k3[4], cx=k3[2], cy=k3[5];
    for (int n = threadIdx.x; n < NTOK; n += 128) {
        float jx = (float)(n & 31), iy = (float)(n >> 5);
        float nc[3] = { (jx - cx)/fx, (iy - cy)/fy, 1.0f };
        float p3[3]; mv3(sr, nc, p3);
        p3[0]+=st[0]; p3[1]+=st[1]; p3[2]+=st[2];
        float p2[3]; mv3(ri, p3, p2);
        p2[0]-=tt[0]; p2[1]-=tt[1]; p2[2]-=tt[2];
        float pu[3]; mv3(k3, p2, pu);
        float pz = pu[2] + 1e-6f;
        float px = pu[0]/pz, py = pu[1]/pz, pw = pu[2]/pz;
        float ox = px - oij0, oy = py - oij1, oz = pw - 1.0f;
        float vlen = sqrtf(ox*ox + oy*oy + oz*oz);
        float* o = opi + ((size_t)(dir * BATCH + b) * NTOK + n) * 4;
        o[0]=ox; o[1]=oy; o[2]=oz; o[3]=1.0f/vlen;
    }
}

// grid = 2*B*N blocks, 128 thr: per-row max of dw
__global__ __launch_bounds__(128) void geom_rowmax(
    const float* __restrict__ opi, const float* __restrict__ oijb,
    float* __restrict__ rowmax)
{
    int idx = blockIdx.x;
    int db = idx >> 10;
    const float* o = oijb + db * 4;
    const float* op = opi + (size_t)idx * 4;
    Op4 opv = { op[0], op[1], op[2], op[3] };
    float mx = -1e30f;
    for (int m = threadIdx.x; m < NTOK; m += 128)
        mx = fmaxf(mx, epi_dw(opv, o[0], o[1], (float)(m & 31), (float)(m >> 5)));
    __shared__ float red[128];
    red[threadIdx.x] = mx; __syncthreads();
    for (int s = 64; s > 0; s >>= 1) {
        if (threadIdx.x < s) red[threadIdx.x] = fmaxf(red[threadIdx.x], red[threadIdx.x + s]);
        __syncthreads();
    }
    if (threadIdx.x == 0) rowmax[idx] = red[0];
}

// grid = B*N blocks: wmap[b,q,k] = e1(q,k) * e2(k,q)
__global__ __launch_bounds__(256) void geom_wmap(
    const float* __restrict__ opi, const float* __restrict__ oijb,
    const float* __restrict__ rowmax, float* __restrict__ wmap)
{
    int b = blockIdx.x >> 10, q = blockIdx.x & 1023;
    const float* o1 = oijb + b * 4;
    const float* o2 = oijb + (BATCH + b) * 4;
    const float* p1 = opi + (size_t)(b * NTOK + q) * 4;
    Op4 op1 = { p1[0], p1[1], p1[2], p1[3] };
    bool flat1 = rowmax[b * NTOK + q] < 0.5f;
    float qx = (float)(q & 31), qy = (float)(q >> 5);
    float* wrow = wmap + ((size_t)b * NTOK + q) * NTOK;
    for (int k = threadIdx.x; k < NTOK; k += 256) {
        float v1 = flat1 ? 1.0f
                         : epi_dw(op1, o1[0], o1[1], (float)(k & 31), (float)(k >> 5));
        const float* p2 = opi + (size_t)((BATCH + b) * NTOK + k) * 4;
        Op4 op2 = { p2[0], p2[1], p2[2], p2[3] };
        float v2 = (rowmax[(BATCH + b) * NTOK + k] < 0.5f)
                       ? 1.0f : epi_dw(op2, o2[0], o2[1], qx, qy);
        wrow[k] = v1 * v2;
    }
}

// broadcast wmap (B,N,N) -> wm_full (B*12,N,N), vectorized b128
__global__ __launch_bounds__(256) void wm_bcast(
    const f32x4* __restrict__ wmap4, f32x4* __restrict__ out4)
{
    const size_t per_src = (size_t)NTOK * (NTOK / 4);
    const size_t per_b   = (size_t)N_HEADS * per_src;
    const size_t total   = (size_t)BATCH * per_b;
    size_t stride = (size_t)gridDim.x * 256;
    for (size_t i = (size_t)blockIdx.x * 256 + threadIdx.x; i < total; i += stride) {
        size_t b = i / per_b;
        size_t src = (i - b * per_b) % per_src;
        out4[i] = wmap4[b * per_src + src];
    }
}

// ---------------- flash attention with wmap-scaled scores --------------------
// grid = B*H*16 blocks, 128 thr (4 waves); each wave owns a 16-row strip.
__global__ __launch_bounds__(128) void attn_kernel(
    const h16* __restrict__ qb, const h16* __restrict__ kb,
    const h16* __restrict__ vtb, const float* __restrict__ wmap,
    h16* __restrict__ ab)
{
    __shared__ h16 qs[64 * 72];
    __shared__ h16 ks[64 * 72];
    __shared__ h16 vs[64 * 72];
    __shared__ h16 ps[4 * 16 * 72];
    int bh = blockIdx.x >> 4, qt = blockIdx.x & 15;
    int b = bh / N_HEADS, h = bh % N_HEADS;
    int tid = threadIdx.x, wave = tid >> 5, lane = tid & 31;
    int hh = lane >> 4, l16 = lane & 15;
    int q0 = qt * 64;
    const float scale = 0.125f;  // 64^-0.5

#pragma unroll
    for (int j = 0; j < 4; ++j) {
        int idx = tid + j * 128;
        int r = idx >> 3, c = idx & 7;
        *(u32x4*)&qs[r * 72 + c * 8] =
            *(const u32x4*)(qb + ((size_t)bh * NTOK + q0 + r) * DHEAD + c * 8);
    }

    v8f acc[4];
    float m_s[8], l_s[8];
#pragma unroll
    for (int t = 0; t < 4; ++t)
#pragma unroll
        for (int r = 0; r < 8; ++r) acc[t][r] = 0.0f;
#pragma unroll
    for (int r = 0; r < 8; ++r) { m_s[r] = -1e30f; l_s[r] = 0.0f; }

    const float* wbase = wmap + (size_t)b * NTOK * NTOK;

    for (int kt = 0; kt < 16; ++kt) {
        int k0 = kt * 64;
#pragma unroll
        for (int j = 0; j < 4; ++j) {
            int idx = tid + j * 128;
            int r = idx >> 3, c = idx & 7;
            *(u32x4*)&ks[r * 72 + c * 8] =
                *(const u32x4*)(kb + ((size_t)bh * NTOK + k0 + r) * DHEAD + c * 8);
            *(u32x4*)&vs[r * 72 + c * 8] =
                *(const u32x4*)(vtb + ((size_t)bh * DHEAD + r) * NTOK + k0 + c * 8);
        }
        __syncthreads();

        // S strip (16x64) = q . k^T
        v8f sf[4];
#pragma unroll
        for (int t = 0; t < 4; ++t)
#pragma unroll
            for (int r = 0; r < 8; ++r) sf[t][r] = 0.0f;
#pragma unroll
        for (int d0 = 0; d0 < 64; d0 += 32) {
            v16h af = lds_a_frag(qs, wave * 16 + l16, 72, d0, hh);
#pragma unroll
            for (int nt = 0; nt < 4; ++nt) {
                v16h bf = lds_b_frag(ks, nt * 16 + l16, 72, d0, hh);
                sf[nt] = __builtin_amdgcn_wmma_f32_16x16x32_f16(
                    false, af, false, bf, (short)0, sf[nt], false, false);
            }
        }

        // scale * wmap, online-softmax stats (half-wave reductions)
        float nm[8];
#pragma unroll
        for (int r = 0; r < 8; ++r) nm[r] = -1e30f;
#pragma unroll
        for (int nt = 0; nt < 4; ++nt) {
            int ck = k0 + nt * 16 + l16;
#pragma unroll
            for (int r = 0; r < 8; ++r) {
                int rq = q0 + wave * 16 + hh * 8 + r;
                float wv = wbase[(size_t)rq * NTOK + ck];
                float sv = sf[nt][r] * scale * wv;
                sf[nt][r] = sv;
                nm[r] = fmaxf(nm[r], sv);
            }
        }
#pragma unroll
        for (int r = 0; r < 8; ++r)
#pragma unroll
            for (int off = 1; off < 16; off <<= 1)
                nm[r] = fmaxf(nm[r], __shfl_xor(nm[r], off, 32));

        float corr[8];
#pragma unroll
        for (int r = 0; r < 8; ++r) {
            float mn = fmaxf(m_s[r], nm[r]);
            corr[r] = __expf(m_s[r] - mn);
            m_s[r] = mn;
        }
        float rs[8];
#pragma unroll
        for (int r = 0; r < 8; ++r) rs[r] = 0.0f;
#pragma unroll
        for (int nt = 0; nt < 4; ++nt)
#pragma unroll
            for (int r = 0; r < 8; ++r) {
                float p = __expf(sf[nt][r] - m_s[r]);
                sf[nt][r] = p;
                rs[r] += p;
            }
#pragma unroll
        for (int r = 0; r < 8; ++r) {
#pragma unroll
            for (int off = 1; off < 16; off <<= 1)
                rs[r] += __shfl_xor(rs[r], off, 32);
            l_s[r] = l_s[r] * corr[r] + rs[r];
#pragma unroll
            for (int nt = 0; nt < 4; ++nt) acc[nt][r] *= corr[r];
        }

        // P (C-layout) -> LDS f16 -> A-layout; wave-private strip
        h16* pst = ps + wave * 16 * 72;
#pragma unroll
        for (int nt = 0; nt < 4; ++nt)
#pragma unroll
            for (int r = 0; r < 8; ++r)
                pst[(hh * 8 + r) * 72 + nt * 16 + l16] = (h16)sf[nt][r];
        asm volatile("s_wait_dscnt 0x0" ::: "memory");

        // O strip += P (16x64) @ V (64x64), B-frags from v^T rows
#pragma unroll
        for (int d0 = 0; d0 < 64; d0 += 32) {
            v16h af = lds_a_frag(pst, l16, 72, d0, hh);
#pragma unroll
            for (int nt = 0; nt < 4; ++nt) {
                v16h bf = lds_b_frag(vs, nt * 16 + l16, 72, d0, hh);
                acc[nt] = __builtin_amdgcn_wmma_f32_16x16x32_f16(
                    false, af, false, bf, (short)0, acc[nt], false, false);
            }
        }
        __syncthreads();
    }

    // normalize, write merged-head layout (B,N,768) as f16 for the proj GEMM
#pragma unroll
    for (int nt = 0; nt < 4; ++nt) {
        int d = nt * 16 + l16;
#pragma unroll
        for (int r = 0; r < 8; ++r) {
            int rq = q0 + wave * 16 + hh * 8 + r;
            float o = acc[nt][r] / l_s[r];
            ab[((size_t)b * NTOK + rq) * EDIM + h * DHEAD + d] = (h16)o;
        }
    }
}

// ---------------- host orchestration -----------------------------------------
extern "C" void kernel_launch(void* const* d_in, const int* in_sizes, int n_in,
                              void* d_out, int out_size, void* d_ws, size_t ws_size,
                              hipStream_t stream) {
    const float* x      = (const float*)d_in[0];
    const float* src    = (const float*)d_in[1];
    const float* intr   = (const float*)d_in[2];
    const float* c2w    = (const float*)d_in[3];
    const float* lnq_g  = (const float*)d_in[4];
    const float* lnq_b  = (const float*)d_in[5];
    const float* Wq     = (const float*)d_in[6];
    const float* bq     = (const float*)d_in[7];
    const float* lnk_g  = (const float*)d_in[8];
    const float* lnk_b  = (const float*)d_in[9];
    const float* Wk     = (const float*)d_in[10];
    const float* bk     = (const float*)d_in[11];
    const float* lnv_g  = (const float*)d_in[12];
    const float* lnv_b  = (const float*)d_in[13];
    const float* Wv     = (const float*)d_in[14];
    const float* bv     = (const float*)d_in[15];
    const float* Wp     = (const float*)d_in[16];
    const float* bp     = (const float*)d_in[17];
    const float* pre_g  = (const float*)d_in[18];
    const float* pre_b  = (const float*)d_in[19];
    const float* W1     = (const float*)d_in[20];
    const float* b1     = (const float*)d_in[21];
    const float* W2     = (const float*)d_in[22];
    const float* b2     = (const float*)d_in[23];
    const float* post_g = (const float*)d_in[24];
    const float* post_b = (const float*)d_in[25];
    float* out = (float*)d_out;
    char* w = (char*)d_ws;

    size_t off = 0;
    auto alloc = [&](size_t bytes) { size_t o = off; off += (bytes + 255) & ~(size_t)255; return o; };
    const size_t actH = (size_t)MROWS * EDIM * 2;       // 6291456
    size_t o_lnq = alloc(actH);
    size_t o_lnk = alloc(actH);
    size_t o_lnv = alloc(actH);
    size_t o_wqt = alloc((size_t)EDIM * EDIM * 2);
    size_t o_wkt = alloc((size_t)EDIM * EDIM * 2);
    size_t o_wvt = alloc((size_t)EDIM * EDIM * 2);
    size_t o_wpt = alloc((size_t)EDIM * EDIM * 2);
    size_t o_w1t = alloc((size_t)EDIM * FFDIM * 2);
    size_t o_w2t = alloc((size_t)EDIM * FFDIM * 2);
    size_t o_q   = alloc(actH);
    size_t o_k   = alloc(actH);
    size_t o_vt  = alloc(actH);
    size_t o_z1  = alloc((size_t)MROWS * EDIM * 4);
    size_t o_zf  = alloc((size_t)MROWS * EDIM * 4);
    size_t o_zh  = alloc(actH);
    size_t o_wm  = alloc((size_t)BATCH * NTOK * NTOK * 4);
    size_t o_opi = alloc(2 * BATCH * NTOK * 4 * 4);
    size_t o_oij = alloc(2 * BATCH * 4 * 4);
    size_t o_rmx = alloc(2 * BATCH * NTOK * 4);
    size_t o_a   = o_lnq;   // alias: ln inputs dead after q/k/v GEMMs
    size_t o_h1  = o_lnk;   // alias: spans lnk+lnv (12.6 MB for 4096x1536 f16)
    size_t o_z2  = o_z1;    // alias: z1 dead after pre-LN

    // 1) weights -> f16 transposed
    transpose_w<<<dim3(EDIM / 32, EDIM / 32), 256, 0, stream>>>(Wq, (h16*)(w + o_wqt), EDIM, EDIM);
    transpose_w<<<dim3(EDIM / 32, EDIM / 32), 256, 0, stream>>>(Wk, (h16*)(w + o_wkt), EDIM, EDIM);
    transpose_w<<<dim3(EDIM / 32, EDIM / 32), 256, 0, stream>>>(Wv, (h16*)(w + o_wvt), EDIM, EDIM);
    transpose_w<<<dim3(EDIM / 32, EDIM / 32), 256, 0, stream>>>(Wp, (h16*)(w + o_wpt), EDIM, EDIM);
    transpose_w<<<dim3(EDIM / 32, FFDIM / 32), 256, 0, stream>>>(W1, (h16*)(w + o_w1t), EDIM, FFDIM);
    transpose_w<<<dim3(FFDIM / 32, EDIM / 32), 256, 0, stream>>>(W2, (h16*)(w + o_w2t), FFDIM, EDIM);

    // 2) input layernorms (f16 out)
    ln_rows<<<MROWS, 256, 0, stream>>>(x,   lnq_g, lnq_b, nullptr, (h16*)(w + o_lnq), EDIM);
    ln_rows<<<MROWS, 256, 0, stream>>>(src, lnk_g, lnk_b, nullptr, (h16*)(w + o_lnk), EDIM);
    ln_rows<<<MROWS, 256, 0, stream>>>(src, lnv_g, lnv_b, nullptr, (h16*)(w + o_lnv), EDIM);

    // 3) q/k/v projections (WMMA)
    dim3 gP(MROWS / 64, EDIM / 128);
    gemm_ws<<<gP, 256, 0, stream>>>((const h16*)(w + o_lnq), (const h16*)(w + o_wqt),
                                    bq, nullptr, w + o_q, EDIM, EDIM, EPI_SPLIT_QK);
    gemm_ws<<<gP, 256, 0, stream>>>((const h16*)(w + o_lnk), (const h16*)(w + o_wkt),
                                    bk, nullptr, w + o_k, EDIM, EDIM, EPI_SPLIT_QK);
    gemm_ws<<<gP, 256, 0, stream>>>((const h16*)(w + o_lnv), (const h16*)(w + o_wvt),
                                    bv, nullptr, w + o_vt, EDIM, EDIM, EPI_SPLIT_VT);

    // 4) epipolar geometry -> wmap, broadcast to d_out
    geom_prep<<<2 * BATCH, 128, 0, stream>>>(intr, c2w, (float*)(w + o_opi), (float*)(w + o_oij));
    geom_rowmax<<<2 * BATCH * NTOK, 128, 0, stream>>>((const float*)(w + o_opi),
                                                      (const float*)(w + o_oij),
                                                      (float*)(w + o_rmx));
    geom_wmap<<<BATCH * NTOK, 256, 0, stream>>>((const float*)(w + o_opi),
                                                (const float*)(w + o_oij),
                                                (const float*)(w + o_rmx),
                                                (float*)(w + o_wm));
    wm_bcast<<<8192, 256, 0, stream>>>((const f32x4*)(w + o_wm),
                                       (f32x4*)(out + (size_t)MROWS * EDIM));

    // 5) attention (flash-style, WMMA both matmuls)
    attn_kernel<<<BATCH * N_HEADS * 16, 128, 0, stream>>>(
        (const h16*)(w + o_q), (const h16*)(w + o_k), (const h16*)(w + o_vt),
        (const float*)(w + o_wm), (h16*)(w + o_a));

    // 6) output projection -> z1 (f32)
    gemm_ws<<<gP, 256, 0, stream>>>((const h16*)(w + o_a), (const h16*)(w + o_wpt),
                                    bp, nullptr, w + o_z1, EDIM, EDIM, EPI_F32);
    // 7) pre-LN: z (f32 + f16)
    ln_rows<<<MROWS, 256, 0, stream>>>((const float*)(w + o_z1), pre_g, pre_b,
                                       (float*)(w + o_zf), (h16*)(w + o_zh), EDIM);
    // 8) FFN1 + gelu (f16)
    gemm_ws<<<dim3(MROWS / 64, FFDIM / 128), 256, 0, stream>>>(
        (const h16*)(w + o_zh), (const h16*)(w + o_w1t),
        b1, nullptr, w + o_h1, EDIM, FFDIM, EPI_GELU16);
    // 9) FFN2 + residual (f32)
    gemm_ws<<<gP, 256, 0, stream>>>((const h16*)(w + o_h1), (const h16*)(w + o_w2t),
                                    b2, (const float*)(w + o_zf), w + o_z2,
                                    FFDIM, EDIM, EPI_RES32);
    // 10) post-LN -> d_out
    ln_rows<<<MROWS, 256, 0, stream>>>((const float*)(w + o_z2), post_g, post_b,
                                       out, nullptr, EDIM);
    (void)in_sizes; (void)n_in; (void)out_size; (void)ws_size;
}
